// LambdaGW_21912923144300
// MI455X (gfx1250) — compile-verified, matrix operand order checked by reading, not run on hardware
//
#include <hip/hip_runtime.h>

#define BATCH 16
#define T1 2048
#define T2 2048
#define TPAD (T1 + 4)
#define IDIM 80
#define IDIMP 96
#define LDIM 64
#define NLAY 4
#define NCOMPS 4
#define KSZ 5
#define CH 256
#define EPSF 1e-5f

typedef __attribute__((ext_vector_type(16))) __bf16 bf16x16;
typedef __attribute__((ext_vector_type(8)))  __bf16 bf16x8;
typedef __attribute__((ext_vector_type(8)))  float  floatx8;

__device__ __forceinline__ floatx8 wmma_bf(bf16x16 a, bf16x16 b, floatx8 c) {
  return __builtin_amdgcn_wmma_f32_16x16x32_bf16(false, a, false, b, (short)0, c, false, false);
}

// ---- WMMA fragment loaders (CDNA5 16x16x32 bf16 layouts, wave32) ----
// A: 16(M)x32(K) row-major. Per lane: M = lane&15; K-half base h = (lane>>4)*8.
// Elements 0..7  <- K = h..h+7       (16 contiguous bytes -> one b128 load)
// Elements 8..15 <- K = 16+h..16+h+7 (16 contiguous bytes -> one b128 load)
__device__ __forceinline__ bf16x16 ldA(const __bf16* p, int ld, int lane) {
  int m = lane & 15, h = (lane >> 4) << 3;
  const __bf16* q = p + m * ld + h;
  bf16x8 lo = *(const bf16x8*)(q);
  bf16x8 hi = *(const bf16x8*)(q + 16);
  return __builtin_shufflevector(lo, hi, 0, 1, 2, 3, 4, 5, 6, 7,
                                 8, 9, 10, 11, 12, 13, 14, 15);
}
// B for X @ W^T: W is (N x K) row-major, K contiguous; B[k][n] = W[n*ld + k].
// Identical lane pattern with N in place of M.
__device__ __forceinline__ bf16x16 ldBt(const __bf16* w, int ld, int lane) {
  int n = lane & 15, h = (lane >> 4) << 3;
  const __bf16* q = w + n * ld + h;
  bf16x8 lo = *(const bf16x8*)(q);
  bf16x8 hi = *(const bf16x8*)(q + 16);
  return __builtin_shufflevector(lo, hi, 0, 1, 2, 3, 4, 5, 6, 7,
                                 8, 9, 10, 11, 12, 13, 14, 15);
}
// C/D element e (0..7): m = e + 8*(lane>>4), n = lane&15

// ---------------- weight conversion / repack kernels ----------------
__global__ void cvt_pad(const float* __restrict__ src, __bf16* __restrict__ dst,
                        int N, int K, int NP, int KP) {
  int i = blockIdx.x * 256 + threadIdx.x;
  if (i >= NP * KP) return;
  int n = i / KP, k = i % KP;
  dst[i] = (__bf16)((n < N && k < K) ? src[n * K + k] : 0.f);
}

// conv W (CH, ic, KSZ) f32  ->  [KSZ][CH][icP] bf16 (K-contiguous for ldBt)
__global__ void cvt_convw(const float* __restrict__ W, __bf16* __restrict__ dst,
                          int ic, int icP) {
  int i = blockIdx.x * 256 + threadIdx.x;
  if (i >= KSZ * CH * icP) return;
  int icp = i % icP, oc = (i / icP) % CH, k = i / (icP * CH);
  dst[i] = (__bf16)((icp < ic) ? W[(oc * ic + icp) * KSZ + k] : 0.f);
}

// ---------------- front end: x + pos_enc -> padded bf16 (b,t,c) ----------------
__global__ void prep_kernel(const float* __restrict__ x, __bf16* __restrict__ h) {
  int i = blockIdx.x * 256 + threadIdx.x;
  if (i >= BATCH * T1) return;
  int b = i >> 11, t = i & 2047;
  const float* xr = x + (size_t)i * IDIM;
  __bf16* hr = h + (size_t)(b * TPAD + t + 2) * CH;
  float tf = (float)t;
#pragma unroll 4
  for (int j = 0; j < 40; ++j) {
    float dv = __expf(-0.09780057519f * (float)(2 * j));
    float a = tf * dv * 4.f;
    hr[j]      = (__bf16)(xr[j] + __sinf(a) * 0.15811388f);
    hr[40 + j] = (__bf16)(xr[40 + j] + __cosf(a) * 0.15811388f);
  }
}

// ---------------- conv layer: staged A halo + 5 shifted GEMMs + bias+relu+LN ----------------
__global__ void __launch_bounds__(256) conv_kernel(
    const __bf16* __restrict__ in, __bf16* __restrict__ out,
    const __bf16* __restrict__ wk,  // [KSZ][CH][icP]
    const float* __restrict__ bias, const float* __restrict__ g,
    const float* __restrict__ be, int kIters, int icP) {
  __shared__ __attribute__((aligned(16))) __bf16 aT[20 * CH];  // 20-row halo tile
  __shared__ float sc[16][CH];
  __shared__ float redS[256], redQ[256];
  __shared__ float stat[16][2];
  int tid = threadIdx.x, lane = tid & 31, wave = tid >> 5;
  int bt = blockIdx.x;
  int b = bt >> 7;
  int tt = (bt & 127) * 16;
  // stage rows [tt .. tt+19] of padded input (all taps) into LDS
  {
    const uint4* s4 = (const uint4*)(in + (size_t)(b * TPAD + tt) * CH);
    uint4* d4 = (uint4*)aT;
    for (int i = tid; i < 20 * CH * 2 / 16; i += 256) d4[i] = s4[i];
  }
  __syncthreads();
  int n0 = wave * 32;
  floatx8 c0 = {}, c1 = {};
  for (int k = 0; k < KSZ; ++k) {
    const __bf16* a = aT + k * CH;
    const __bf16* w = wk + (size_t)k * CH * icP;
    for (int ki = 0; ki < kIters; ++ki) {
      bf16x16 af = ldA(a + ki * 32, CH, lane);
      bf16x16 b0 = ldBt(w + (size_t)n0 * icP + ki * 32, icP, lane);
      bf16x16 b1 = ldBt(w + (size_t)(n0 + 16) * icP + ki * 32, icP, lane);
      c0 = wmma_bf(af, b0, c0);
      c1 = wmma_bf(af, b1, c1);
    }
  }
  int nc = lane & 15, mh = (lane >> 4) << 3;
#pragma unroll
  for (int e = 0; e < 8; ++e) {
    int m = e + mh;
    sc[m][n0 + nc]      = fmaxf(c0[e] + bias[n0 + nc], 0.f);
    sc[m][n0 + 16 + nc] = fmaxf(c1[e] + bias[n0 + 16 + nc], 0.f);
  }
  __syncthreads();
  int r = tid >> 4, j = tid & 15;
  float s = 0.f, q = 0.f;
#pragma unroll
  for (int t = 0; t < 16; ++t) { float v = sc[r][j + 16 * t]; s += v; q += v * v; }
  redS[tid] = s; redQ[tid] = q;
  __syncthreads();
  if (j == 0) {
    float ts = 0.f, tq = 0.f;
    for (int t = 0; t < 16; ++t) { ts += redS[(r << 4) + t]; tq += redQ[(r << 4) + t]; }
    float mean = ts * (1.f / CH);
    float var = tq * (1.f / CH) - mean * mean;
    stat[r][0] = mean; stat[r][1] = rsqrtf(var + EPSF);
  }
  __syncthreads();
  float mean = stat[r][0], rinv = stat[r][1];
  __bf16* orow = out + (size_t)(b * TPAD + tt + r + 2) * CH;
  for (int t = 0; t < 16; ++t) {
    int col = j + 16 * t;
    orow[col] = (__bf16)((sc[r][col] - mean) * rinv * g[col] + be[col]);
  }
}

// ---------------- k/v projection; k -> (b,t,64) f32, v -> transposed (b,96,T1) bf16 ----------------
__global__ void __launch_bounds__(320) kv_kernel(
    const __bf16* __restrict__ h, const __bf16* __restrict__ wkb,
    const __bf16* __restrict__ wvb, const float* __restrict__ bk,
    const float* __restrict__ bv, const int* __restrict__ ilens,
    float* __restrict__ kout, __bf16* __restrict__ voutT) {
  __shared__ __attribute__((aligned(16))) __bf16 aT[16 * CH];
  int tid = threadIdx.x, lane = tid & 31, wave = tid >> 5;  // 10 waves
  int bt = blockIdx.x;
  int b = bt >> 7, tt = (bt & 127) * 16;
  {
    const uint4* s4 = (const uint4*)(h + (size_t)(b * TPAD + tt + 2) * CH);
    uint4* d4 = (uint4*)aT;
    for (int i = tid; i < 16 * CH * 2 / 16; i += 320) d4[i] = s4[i];
  }
  __syncthreads();
  bool isK = (wave < 4);
  int n0 = isK ? wave * 16 : (wave - 4) * 16;
  const __bf16* w = (isK ? wkb : wvb) + (size_t)n0 * CH;
  floatx8 c = {};
  for (int ki = 0; ki < 8; ++ki) {
    bf16x16 af = ldA(aT + ki * 32, CH, lane);
    bf16x16 bf = ldBt(w + ki * 32, CH, lane);
    c = wmma_bf(af, bf, c);
  }
  int nc = lane & 15, mh = (lane >> 4) << 3;
  int ilen = ilens[b];
#pragma unroll
  for (int e = 0; e < 8; ++e) {
    int t = tt + e + mh, n = n0 + nc;
    if (isK) {
      kout[((size_t)b * T1 + t) * LDIM + n] = c[e] + bk[n];
    } else {
      float v = (n < IDIM) ? (c[e] + bv[n]) : 0.f;
      if (t >= ilen) v = 0.f;
      voutT[((size_t)b * IDIMP + n) * T1 + t] = (__bf16)v;
    }
  }
}

// ---------------- softmax over time -> transposed scores (b,64,T1) bf16 ----------------
__global__ void __launch_bounds__(256) softmax_kernel(
    const float* __restrict__ kbuf, const int* __restrict__ ilens,
    __bf16* __restrict__ soutT) {
  __shared__ float red[256];
  int bl = blockIdx.x;
  int b = bl >> 6, l = bl & 63;
  int tid = threadIdx.x;
  int ilen = ilens[b];
  const float* kp = kbuf + (size_t)b * T1 * LDIM + l;
  float mx = -3.4e38f;
  for (int t = tid; t < T1; t += 256)
    if (t < ilen) mx = fmaxf(mx, kp[(size_t)t * LDIM]);
  red[tid] = mx; __syncthreads();
  for (int s = 128; s > 0; s >>= 1) {
    if (tid < s) red[tid] = fmaxf(red[tid], red[tid + s]);
    __syncthreads();
  }
  mx = red[0]; __syncthreads();
  float sum = 0.f;
  for (int t = tid; t < T1; t += 256)
    if (t < ilen) sum += __expf(kp[(size_t)t * LDIM] - mx);
  red[tid] = sum; __syncthreads();
  for (int s = 128; s > 0; s >>= 1) {
    if (tid < s) red[tid] += red[tid + s];
    __syncthreads();
  }
  float inv = 1.f / red[0];
  __bf16* orow = soutT + ((size_t)b * LDIM + l) * T1;
  for (int t = tid; t < T1; t += 256) {
    float v = (t < ilen) ? __expf(kp[(size_t)t * LDIM] - mx) * inv : 0.f;
    orow[t] = (__bf16)v;
  }
}

// ---------------- l = scores^T @ v  (64x2048 @ 2048x96, both K-contiguous) ----------------
__global__ void __launch_bounds__(32) lmat_kernel(
    const __bf16* __restrict__ scT, const __bf16* __restrict__ vbT,
    __bf16* __restrict__ lbf) {
  int lane = threadIdx.x & 31;
  int b = blockIdx.x;
  int mt = blockIdx.y / 6, nt = blockIdx.y % 6;
  const __bf16* abase = scT + ((size_t)b * LDIM + mt * 16) * T1;  // A[m][k] = scT[m*T1+k]
  const __bf16* bbase = vbT + ((size_t)b * IDIMP + nt * 16) * T1; // B[k][n] = vbT[n*T1+k]
  floatx8 c = {};
  for (int ki = 0; ki < T1 / 32; ++ki) {
    bf16x16 af = ldA(abase + ki * 32, T1, lane);
    bf16x16 bf = ldBt(bbase + ki * 32, T1, lane);
    c = wmma_bf(af, bf, c);
  }
  int nc = lane & 15, mh = (lane >> 4) << 3;
#pragma unroll
  for (int e = 0; e < 8; ++e)
    lbf[((size_t)b * 64 + mt * 16 + e + mh) * IDIMP + nt * 16 + nc] = (__bf16)c[e];
}

// ---------------- lf layer: staged A + GEMM + bias + relu + LN(256) ----------------
__global__ void __launch_bounds__(256) lf_kernel(
    const __bf16* __restrict__ in, int ldin, int kIters,
    const __bf16* __restrict__ w, int ldw, const float* __restrict__ bias,
    const float* __restrict__ g, const float* __restrict__ be,
    __bf16* __restrict__ out) {
  __shared__ __attribute__((aligned(16))) __bf16 aT[16 * CH];
  __shared__ float sc[16][CH];
  __shared__ float redS[256], redQ[256];
  __shared__ float stat[16][2];
  int tid = threadIdx.x, lane = tid & 31, wave = tid >> 5;
  int rt = blockIdx.x * 16;
  {
    const uint4* s4 = (const uint4*)(in + (size_t)rt * ldin);
    uint4* d4 = (uint4*)aT;
    for (int i = tid; i < 16 * ldin * 2 / 16; i += 256) d4[i] = s4[i];
  }
  __syncthreads();
  int n0 = wave * 32;
  floatx8 c0 = {}, c1 = {};
  for (int ki = 0; ki < kIters; ++ki) {
    bf16x16 af = ldA(aT + ki * 32, ldin, lane);
    bf16x16 b0 = ldBt(w + (size_t)n0 * ldw + ki * 32, ldw, lane);
    bf16x16 b1 = ldBt(w + (size_t)(n0 + 16) * ldw + ki * 32, ldw, lane);
    c0 = wmma_bf(af, b0, c0);
    c1 = wmma_bf(af, b1, c1);
  }
  int nc = lane & 15, mh = (lane >> 4) << 3;
#pragma unroll
  for (int e = 0; e < 8; ++e) {
    int m = e + mh;
    sc[m][n0 + nc]      = fmaxf(c0[e] + bias[n0 + nc], 0.f);
    sc[m][n0 + 16 + nc] = fmaxf(c1[e] + bias[n0 + 16 + nc], 0.f);
  }
  __syncthreads();
  int r = tid >> 4, j = tid & 15;
  float s = 0.f, q = 0.f;
#pragma unroll
  for (int t = 0; t < 16; ++t) { float v = sc[r][j + 16 * t]; s += v; q += v * v; }
  redS[tid] = s; redQ[tid] = q;
  __syncthreads();
  if (j == 0) {
    float ts = 0.f, tq = 0.f;
    for (int t = 0; t < 16; ++t) { ts += redS[(r << 4) + t]; tq += redQ[(r << 4) + t]; }
    float mean = ts * (1.f / CH);
    float var = tq * (1.f / CH) - mean * mean;
    stat[r][0] = mean; stat[r][1] = rsqrtf(var + EPSF);
  }
  __syncthreads();
  float mean = stat[r][0], rinv = stat[r][1];
  __bf16* orow = out + (size_t)(rt + r) * CH;
  for (int t = 0; t < 16; ++t) {
    int col = j + 16 * t;
    orow[col] = (__bf16)((sc[r][col] - mean) * rinv * g[col] + be[col]);
  }
}

// ---------------- lf_out: 1024x256 @ 256x1024 + bias -> f32 ----------------
__global__ void __launch_bounds__(256) lfout_kernel(
    const __bf16* __restrict__ in, const __bf16* __restrict__ w,
    const float* __restrict__ bias, float* __restrict__ out) {
  __shared__ __attribute__((aligned(16))) __bf16 aT[16 * CH];
  int tid = threadIdx.x, lane = tid & 31, wave = tid >> 5;
  int rt = blockIdx.x * 16;
  {
    const uint4* s4 = (const uint4*)(in + (size_t)rt * CH);
    uint4* d4 = (uint4*)aT;
    for (int i = tid; i < 16 * CH * 2 / 16; i += 256) d4[i] = s4[i];
  }
  __syncthreads();
  int n0 = (blockIdx.y * 8 + wave) * 16;
  floatx8 c = {};
  for (int ki = 0; ki < 8; ++ki) {
    bf16x16 af = ldA(aT + ki * 32, CH, lane);
    bf16x16 bf = ldBt(w + (size_t)n0 * CH + ki * 32, CH, lane);
    c = wmma_bf(af, bf, c);
  }
  int nc = lane & 15, mh = (lane >> 4) << 3;
#pragma unroll
  for (int e = 0; e < 8; ++e)
    out[(size_t)(rt + e + mh) * 1024 + n0 + nc] = c[e] + bias[n0 + nc];
}

// lfout (b,64,1024) -> Lmats [comp][b][layer][m][l] bf16  (transposed: K=l contiguous)
__global__ void repack_l(const float* __restrict__ lfout, __bf16* __restrict__ Lm) {
  int i = blockIdx.x * 256 + threadIdx.x;
  if (i >= NCOMPS * BATCH * NLAY * 64 * 64) return;
  int l = i & 63, m = (i >> 6) & 63, lay = (i >> 12) & 3, b = (i >> 14) & 15, j = (i >> 18) & 3;
  Lm[i] = (__bf16)lfout[((size_t)(b * 64 + l)) * 1024 + m * 16 + lay * 4 + j];
}

__global__ void finit_kernel(const int* __restrict__ il, const int* __restrict__ ol,
                             float* __restrict__ f) {
  int i = blockIdx.x * 256 + threadIdx.x;
  if (i >= BATCH * T2) return;
  int b = i >> 11, t = i & 2047;
  f[i] = (float)t * ((float)il[b] / (float)ol[b]);
}

__global__ void rk4_combine(float* __restrict__ f, const float* __restrict__ k1,
                            const float* __restrict__ k2, const float* __restrict__ k3,
                            const float* __restrict__ k4) {
  int i = blockIdx.x * 256 + threadIdx.x;
  if (i >= BATCH * T2) return;
  f[i] += (k1[i] + 2.f * k2[i] + 2.f * k3[i] + k4[i]) * (1.f / 6.f);
}

// ---------------- fused gw_signal ----------------
struct GWParams {
  const __bf16* Wemb;            // 64 x 96 (K-padded)
  const float* bemb;
  const float* gs; const float* bs;
  const __bf16* W1[4];           // 256 x 64
  const float* b1[4]; const float* gl[4]; const float* bt[4];
  const __bf16* W2[4];           // 64x256 (layer3: 16x256 row-padded)
  const float* b2[4];
  const __bf16* Lm;              // [B][4][m=64][l=64] (transposed) for current comp
};

#define GW_SMEM 145920

__global__ void __launch_bounds__(256) gw_kernel(
    const float* __restrict__ fbase, const float* __restrict__ kprev, float alpha,
    GWParams P, float* __restrict__ kout) {
  extern __shared__ __attribute__((aligned(16))) char smem[];
  __bf16* penc = (__bf16*)smem;                          // 64*96 bf16  = 12288
  float*  sF   = (float*)(smem + 12288);                 // 64*64 f32   = 16384
  __bf16* sB   = (__bf16*)(smem + 12288 + 16384);        // 64*64 bf16  = 8192
  __bf16* snB  = (__bf16*)(smem + 12288 + 16384 + 8192); // 64*64 bf16  = 8192
  float*  uF   = (float*)(smem + 45056);                 // 64*256 f32  = 65536
  __bf16* uB   = (__bf16*)(smem + 110592);               // 64*256 bf16 = 32768
  float*  redS = (float*)(smem + 143360);                // 256 f32
  float*  redQ = (float*)(smem + 144384);                // 256 f32
  float*  stat = (float*)(smem + 145408);                // 64*2 f32

  int tid = threadIdx.x, lane = tid & 31, wave = tid >> 5;
  int b = blockIdx.x >> 5;
  int t0 = (blockIdx.x & 31) * 64;

  // positions + positional encoding into LDS (cols 80..95 zeroed)
  {
    int r = tid >> 2, q = tid & 3;
    size_t idx = (size_t)b * T2 + t0 + r;
    float p = fbase[idx] + alpha * kprev[idx];
#pragma unroll 2
    for (int jj = 0; jj < 10; ++jj) {
      int j = q * 10 + jj;
      float dv = __expf(-0.09780057519f * (float)(2 * j));
      float a = p * dv * 4.f;
      penc[r * IDIMP + j]      = (__bf16)(__sinf(a) * 0.15811388f);
      penc[r * IDIMP + 40 + j] = (__bf16)(__cosf(a) * 0.15811388f);
    }
#pragma unroll
    for (int z = 0; z < 4; ++z) penc[r * IDIMP + 80 + q * 4 + z] = (__bf16)0.f;
  }
  __syncthreads();

  // s = penc @ Wemb^T + bemb   (64x96 @ 96x64)
  for (int sIt = 0; sIt < 2; ++sIt) {
    int tIdx = wave * 2 + sIt;
    int m0 = (tIdx >> 2) * 16, n0 = (tIdx & 3) * 16;
    floatx8 c = {};
    for (int ki = 0; ki < 3; ++ki) {
      bf16x16 af = ldA(penc + m0 * IDIMP + ki * 32, IDIMP, lane);
      bf16x16 bf = ldBt(P.Wemb + n0 * IDIMP + ki * 32, IDIMP, lane);
      c = wmma_bf(af, bf, c);
    }
    int nc = lane & 15, mh = (lane >> 4) << 3;
#pragma unroll
    for (int e = 0; e < 8; ++e)
      sF[(m0 + e + mh) * LDIM + n0 + nc] = c[e] + P.bemb[n0 + nc];
  }
  __syncthreads();

  for (int L = 0; L < NLAY; ++L) {
    // LN over 64 (gs, bs) -> snB (bf16)
    {
      int r = tid >> 2, q = tid & 3;
      float s = 0.f, qq = 0.f;
#pragma unroll
      for (int c = 0; c < 16; ++c) { float v = sF[r * LDIM + q * 16 + c]; s += v; qq += v * v; }
      redS[tid] = s; redQ[tid] = qq;
      __syncthreads();
      if (q == 0) {
        float ts = redS[r * 4] + redS[r * 4 + 1] + redS[r * 4 + 2] + redS[r * 4 + 3];
        float tq = redQ[r * 4] + redQ[r * 4 + 1] + redQ[r * 4 + 2] + redQ[r * 4 + 3];
        float mean = ts * (1.f / 64.f);
        float var = tq * (1.f / 64.f) - mean * mean;
        stat[r * 2] = mean; stat[r * 2 + 1] = rsqrtf(var + EPSF);
      }
      __syncthreads();
      float mean = stat[r * 2], rv = stat[r * 2 + 1];
#pragma unroll
      for (int c = 0; c < 16; ++c) {
        int col = q * 16 + c;
        snB[r * LDIM + col] = (__bf16)((sF[r * LDIM + col] - mean) * rv * P.gs[col] + P.bs[col]);
      }
    }
    __syncthreads();

    // s += snB @ Lm[b][L]   (residual; L stored transposed -> ldBt)
    const __bf16* Lmb = P.Lm + ((size_t)(b * NLAY + L)) * 64 * 64;
    for (int sIt = 0; sIt < 2; ++sIt) {
      int tIdx = wave * 2 + sIt;
      int m0 = (tIdx >> 2) * 16, n0 = (tIdx & 3) * 16;
      int nc = lane & 15, mh = (lane >> 4) << 3;
      floatx8 c;
#pragma unroll
      for (int e = 0; e < 8; ++e) c[e] = sF[(m0 + e + mh) * LDIM + n0 + nc];
      for (int ki = 0; ki < 2; ++ki) {
        bf16x16 af = ldA(snB + m0 * LDIM + ki * 32, LDIM, lane);
        bf16x16 bf = ldBt(Lmb + (size_t)n0 * LDIM + ki * 32, LDIM, lane);
        c = wmma_bf(af, bf, c);
      }
#pragma unroll
      for (int e = 0; e < 8; ++e) {
        float v = c[e];
        sF[(m0 + e + mh) * LDIM + n0 + nc] = v;
        sB[(m0 + e + mh) * LDIM + n0 + nc] = (__bf16)v;
      }
    }
    __syncthreads();

    // u = relu(sB @ W1^T + b1)  (64x64 @ 64x256)
    for (int sIt = 0; sIt < 8; ++sIt) {
      int tIdx = wave * 8 + sIt;
      int m0 = (tIdx >> 4) * 16, n0 = (tIdx & 15) * 16;
      floatx8 c = {};
      for (int ki = 0; ki < 2; ++ki) {
        bf16x16 af = ldA(sB + m0 * LDIM + ki * 32, LDIM, lane);
        bf16x16 bf = ldBt(P.W1[L] + (size_t)n0 * LDIM + ki * 32, LDIM, lane);
        c = wmma_bf(af, bf, c);
      }
      int nc = lane & 15, mh = (lane >> 4) << 3;
#pragma unroll
      for (int e = 0; e < 8; ++e)
        uF[(m0 + e + mh) * CH + n0 + nc] = fmaxf(c[e] + P.b1[L][n0 + nc], 0.f);
    }
    __syncthreads();

    // LN over 256 (gl, bt) -> uB (bf16)
    {
      int r = tid >> 2, q = tid & 3;
      float s = 0.f, qq = 0.f;
      for (int c = 0; c < 64; ++c) { float v = uF[r * CH + q * 64 + c]; s += v; qq += v * v; }
      redS[tid] = s; redQ[tid] = qq;
      __syncthreads();
      if (q == 0) {
        float ts = redS[r * 4] + redS[r * 4 + 1] + redS[r * 4 + 2] + redS[r * 4 + 3];
        float tq = redQ[r * 4] + redQ[r * 4 + 1] + redQ[r * 4 + 2] + redQ[r * 4 + 3];
        float mean = ts * (1.f / CH);
        float var = tq * (1.f / CH) - mean * mean;
        stat[r * 2] = mean; stat[r * 2 + 1] = rsqrtf(var + EPSF);
      }
      __syncthreads();
      float mean = stat[r * 2], rv = stat[r * 2 + 1];
      for (int c = 0; c < 64; ++c) {
        int col = q * 64 + c;
        uB[r * CH + col] = (__bf16)((uF[r * CH + col] - mean) * rv * P.gl[L][col] + P.bt[L][col]);
      }
    }
    __syncthreads();

    // s = uB @ W2^T + b2  (64x256 @ 256x{64|16})
    for (int sIt = 0; sIt < 2; ++sIt) {
      int tIdx = wave * 2 + sIt;
      int m0 = (tIdx >> 2) * 16, n0 = (tIdx & 3) * 16;
      if (L == 3 && n0 != 0) continue;
      floatx8 c = {};
      for (int ki = 0; ki < 8; ++ki) {
        bf16x16 af = ldA(uB + m0 * CH + ki * 32, CH, lane);
        bf16x16 bf = ldBt(P.W2[L] + (size_t)n0 * CH + ki * 32, CH, lane);
        c = wmma_bf(af, bf, c);
      }
      int nc = lane & 15, mh = (lane >> 4) << 3;
      float bb = (L == 3) ? ((n0 + nc == 0) ? P.b2[3][0] : 0.f) : P.b2[L][n0 + nc];
#pragma unroll
      for (int e = 0; e < 8; ++e)
        sF[(m0 + e + mh) * LDIM + n0 + nc] = c[e] + bb;
    }
    __syncthreads();
  }

  if (tid < 64) kout[(size_t)b * T2 + t0 + tid] = sF[tid * LDIM] * 0.25f;
}

// ---------------- host orchestration ----------------
extern "C" void kernel_launch(void* const* d_in, const int* in_sizes, int n_in,
                              void* d_out, int out_size, void* d_ws, size_t ws_size,
                              hipStream_t stream) {
  (void)in_sizes; (void)n_in; (void)out_size; (void)ws_size;
  const float* x = (const float*)d_in[0];
  const int* ilens = (const int*)d_in[1];
  const int* olens = (const int*)d_in[2];
  int pi = 3;
  const float *convW[4], *convB[4], *convG[4], *convBe[4];
  for (int i = 0; i < 4; ++i) {
    convW[i] = (const float*)d_in[pi++]; convB[i] = (const float*)d_in[pi++];
    convG[i] = (const float*)d_in[pi++]; convBe[i] = (const float*)d_in[pi++];
  }
  const float *lfW[4], *lfB[4], *lfG[4], *lfBe[4];
  for (int i = 0; i < 4; ++i) {
    lfW[i] = (const float*)d_in[pi++]; lfB[i] = (const float*)d_in[pi++];
    lfG[i] = (const float*)d_in[pi++]; lfBe[i] = (const float*)d_in[pi++];
  }
  const float* lfoW = (const float*)d_in[pi++];
  const float* lfoB = (const float*)d_in[pi++];
  const float* kW = (const float*)d_in[pi++];
  const float* kB = (const float*)d_in[pi++];
  const float* vW = (const float*)d_in[pi++];
  const float* vB = (const float*)d_in[pi++];
  const float* gsP = (const float*)d_in[pi++];
  const float* bsP = (const float*)d_in[pi++];
  const float* embW = (const float*)d_in[pi++];
  const float* embB = (const float*)d_in[pi++];
  const float *sW1[4], *sB1[4], *sG[4], *sBt[4], *sW2[4], *sB2[4];
  for (int i = 0; i < 4; ++i) {
    sW1[i] = (const float*)d_in[pi++]; sB1[i] = (const float*)d_in[pi++];
    sG[i] = (const float*)d_in[pi++]; sBt[i] = (const float*)d_in[pi++];
    sW2[i] = (const float*)d_in[pi++]; sB2[i] = (const float*)d_in[pi++];
  }

  size_t off = 0;
  auto carve = [&](size_t bytes) -> char* {
    char* p = (char*)d_ws + off;
    off += (bytes + 255) & ~(size_t)255;
    return p;
  };
  __bf16* hA = (__bf16*)carve((size_t)BATCH * TPAD * CH * 2);
  __bf16* hB = (__bf16*)carve((size_t)BATCH * TPAD * CH * 2);
  __bf16* wconv[4];
  wconv[0] = (__bf16*)carve((size_t)KSZ * CH * IDIMP * 2);
  for (int i = 1; i < 4; ++i) wconv[i] = (__bf16*)carve((size_t)KSZ * CH * CH * 2);
  __bf16* wkb = (__bf16*)carve((size_t)64 * 256 * 2);
  __bf16* wvb = (__bf16*)carve((size_t)96 * 256 * 2);
  float* kbuf = (float*)carve((size_t)BATCH * T1 * 64 * 4);
  __bf16* vbufT = (__bf16*)carve((size_t)BATCH * IDIMP * T1 * 2);
  __bf16* scoresT = (__bf16*)carve((size_t)BATCH * 64 * T1 * 2);
  __bf16* lbf = (__bf16*)carve((size_t)BATCH * 64 * IDIMP * 2);
  __bf16* wlfb[4];
  wlfb[0] = (__bf16*)carve((size_t)256 * 96 * 2);
  for (int i = 1; i < 4; ++i) wlfb[i] = (__bf16*)carve((size_t)256 * 256 * 2);
  __bf16* actA = (__bf16*)carve((size_t)1024 * 256 * 2);
  __bf16* actB = (__bf16*)carve((size_t)1024 * 256 * 2);
  __bf16* wlfo = (__bf16*)carve((size_t)1024 * 256 * 2);
  float* lfoutBuf = (float*)carve((size_t)BATCH * 64 * 1024 * 4);
  __bf16* Lmats = (__bf16*)carve((size_t)NCOMPS * BATCH * NLAY * 64 * 64 * 2);
  __bf16* wemb = (__bf16*)carve((size_t)64 * 96 * 2);
  __bf16* w1b[4]; for (int i = 0; i < 4; ++i) w1b[i] = (__bf16*)carve((size_t)256 * 64 * 2);
  __bf16* w2b[4];
  for (int i = 0; i < 3; ++i) w2b[i] = (__bf16*)carve((size_t)64 * 256 * 2);
  w2b[3] = (__bf16*)carve((size_t)16 * 256 * 2);
  float* kst[4]; for (int i = 0; i < 4; ++i) kst[i] = (float*)carve((size_t)BATCH * T2 * 4);
  float* fOut = (float*)d_out;

  // zero padded conv buffers (covers time pads and channel pads for layer 0)
  hipMemsetAsync(hA, 0, (size_t)BATCH * TPAD * CH * 2, stream);
  hipMemsetAsync(hB, 0, (size_t)BATCH * TPAD * CH * 2, stream);

  // weight conversions to WMMA-friendly (K-contiguous) bf16 layouts
  cvt_convw<<<(KSZ * CH * IDIMP + 255) / 256, 256, 0, stream>>>(convW[0], wconv[0], IDIM, IDIMP);
  for (int i = 1; i < 4; ++i)
    cvt_convw<<<(KSZ * CH * CH + 255) / 256, 256, 0, stream>>>(convW[i], wconv[i], CH, CH);
  cvt_pad<<<(64 * 256 + 255) / 256, 256, 0, stream>>>(kW, wkb, 64, 256, 64, 256);
  cvt_pad<<<(96 * 256 + 255) / 256, 256, 0, stream>>>(vW, wvb, 80, 256, 96, 256);
  cvt_pad<<<(256 * 96 + 255) / 256, 256, 0, stream>>>(lfW[0], wlfb[0], 256, 80, 256, 96);
  for (int i = 1; i < 4; ++i)
    cvt_pad<<<(256 * 256 + 255) / 256, 256, 0, stream>>>(lfW[i], wlfb[i], 256, 256, 256, 256);
  cvt_pad<<<(1024 * 256 + 255) / 256, 256, 0, stream>>>(lfoW, wlfo, 1024, 256, 1024, 256);
  cvt_pad<<<(64 * 96 + 255) / 256, 256, 0, stream>>>(embW, wemb, 64, 80, 64, 96);
  for (int i = 0; i < 4; ++i)
    cvt_pad<<<(256 * 64 + 255) / 256, 256, 0, stream>>>(sW1[i], w1b[i], 256, 64, 256, 64);
  for (int i = 0; i < 3; ++i)
    cvt_pad<<<(64 * 256 + 255) / 256, 256, 0, stream>>>(sW2[i], w2b[i], 64, 256, 64, 256);
  cvt_pad<<<(16 * 256 + 255) / 256, 256, 0, stream>>>(sW2[3], w2b[3], 1, 256, 16, 256);

  // encoder
  prep_kernel<<<128, 256, 0, stream>>>(x, hA);
  conv_kernel<<<2048, 256, 0, stream>>>(hA, hB, wconv[0], convB[0], convG[0], convBe[0], 3, IDIMP);
  conv_kernel<<<2048, 256, 0, stream>>>(hB, hA, wconv[1], convB[1], convG[1], convBe[1], 8, CH);
  conv_kernel<<<2048, 256, 0, stream>>>(hA, hB, wconv[2], convB[2], convG[2], convBe[2], 8, CH);
  conv_kernel<<<2048, 256, 0, stream>>>(hB, hA, wconv[3], convB[3], convG[3], convBe[3], 8, CH);

  // attention pooling
  kv_kernel<<<2048, 320, 0, stream>>>(hA, wkb, wvb, kB, vB, ilens, kbuf, vbufT);
  softmax_kernel<<<1024, 256, 0, stream>>>(kbuf, ilens, scoresT);
  lmat_kernel<<<dim3(16, 24), 32, 0, stream>>>(scoresT, vbufT, lbf);

  // lambda MLP
  lf_kernel<<<64, 256, 0, stream>>>(lbf, IDIMP, 3, wlfb[0], 96, lfB[0], lfG[0], lfBe[0], actA);
  lf_kernel<<<64, 256, 0, stream>>>(actA, 256, 8, wlfb[1], 256, lfB[1], lfG[1], lfBe[1], actB);
  lf_kernel<<<64, 256, 0, stream>>>(actB, 256, 8, wlfb[2], 256, lfB[2], lfG[2], lfBe[2], actA);
  lf_kernel<<<64, 256, 0, stream>>>(actA, 256, 8, wlfb[3], 256, lfB[3], lfG[3], lfBe[3], actB);
  lfout_kernel<<<dim3(64, 8), 256, 0, stream>>>(actB, wlfo, lfoB, lfoutBuf);
  repack_l<<<4096, 256, 0, stream>>>(lfoutBuf, Lmats);

  // RK4 integration
  finit_kernel<<<128, 256, 0, stream>>>(ilens, olens, fOut);

  hipFuncSetAttribute((const void*)gw_kernel,
                      hipFuncAttributeMaxDynamicSharedMemorySize, GW_SMEM);

  GWParams P;
  P.Wemb = wemb; P.bemb = embB; P.gs = gsP; P.bs = bsP;
  for (int i = 0; i < 4; ++i) {
    P.W1[i] = w1b[i]; P.b1[i] = sB1[i]; P.gl[i] = sG[i]; P.bt[i] = sBt[i];
    P.W2[i] = w2b[i]; P.b2[i] = sB2[i];
  }
  for (int j = 0; j < NCOMPS; ++j) {
    P.Lm = Lmats + (size_t)j * BATCH * NLAY * 64 * 64;
    gw_kernel<<<512, 256, GW_SMEM, stream>>>(fOut, fOut, 0.0f, P, kst[0]);
    gw_kernel<<<512, 256, GW_SMEM, stream>>>(fOut, kst[0], 0.5f, P, kst[1]);
    gw_kernel<<<512, 256, GW_SMEM, stream>>>(fOut, kst[1], 0.5f, P, kst[2]);
    gw_kernel<<<512, 256, GW_SMEM, stream>>>(fOut, kst[2], 1.0f, P, kst[3]);
    rk4_combine<<<128, 256, 0, stream>>>(fOut, kst[0], kst[1], kst[2], kst[3]);
  }
}
